// GPT_5557687681273
// MI455X (gfx1250) — matrix operand order repaired and checked by reading, not compiled, not run on hardware
//
#include <hip/hip_runtime.h>
#include <hip/hip_bf16.h>

// ---------------------------------------------------------------------------
// GPT forward for MI455X (gfx1250).
// - All GEMMs on the matrix pipe: v_wmma_f32_16x16x32_f16.
// - GEMM A/B tiles staged into LDS by the Tensor Data Mover
//   (tensor_load_to_lds, TENSORcnt, double-buffered; D# pad fields reproduce
//   the conflict-free 72-half LDS stride), waves only read fragments + WMMA.
// - Flash attention: V pre-transposed by the QKV GEMM epilogue; P row-sums
//   computed on the matrix pipe (WMMA vs ones-fragment).
// ---------------------------------------------------------------------------

typedef __attribute__((ext_vector_type(16))) _Float16 v16h;
typedef __attribute__((ext_vector_type(8)))  float    v8f;
typedef __attribute__((ext_vector_type(4)))  unsigned v4u;
typedef __attribute__((ext_vector_type(8)))  int      v8i;
typedef __attribute__((ext_vector_type(4)))  int      v4i;

#define NL   6
#define NH   16
#define DM   1024
#define HD   64
#define NV   32000
#define NT   1024
#define NB   2
#define DFF  4096
#define NTOK (NB * NT)      // 2048 rows in the token dimension

__device__ __forceinline__ v8f wmma_f16(v16h a, v16h b, v8f c) {
  return __builtin_amdgcn_wmma_f32_16x16x32_f16(
      /*neg_a=*/false, a, /*neg_b=*/false, b,
      /*c_mod=*/(short)0, c, /*reuse_a=*/false, /*reuse_b=*/false);
}

// 16 contiguous halves -> one fragment register set (2 x b128)
__device__ __forceinline__ v16h ld16(const _Float16* p) {
  union { v16h v; uint4 u[2]; } t;
  t.u[0] = *(const uint4*)(p);
  t.u[1] = *(const uint4*)(p + 8);
  return t.v;
}
// two 8-half chunks (A-fragment: K = 16*hf..16*hf+7 and 16*hf+8..16*hf+15)
__device__ __forceinline__ v16h ld8x2(const _Float16* p0, const _Float16* p1) {
  union { v16h v; uint4 u[2]; } t;
  t.u[0] = *(const uint4*)(p0);
  t.u[1] = *(const uint4*)(p1);
  return t.v;
}

// LDS byte offset of a __shared__ object (addrspace(3) numeric address)
__device__ __forceinline__ unsigned lds_off(const void* p) {
  return (unsigned)(unsigned long long)
      (__attribute__((address_space(3))) const void*)p;
}

// ---------------------------------------------------------------------------
// TDM: 2-D tile load Global -> LDS.
// Tile: rows x 64 halves (data_size=2B), row stride K elements in memory.
// LDS: rows are padded to 72 halves via D# padding:
//   pad_interval=4 (insert pad every 32 DWORDs = 64 halves),
//   pad_amount=3   (pad 4 DWORDs = 8 halves)  -> 144 B row stride.
// ---------------------------------------------------------------------------
__device__ __forceinline__ void tdm_load_tile(const _Float16* gptr,
                                              unsigned lds_addr,
                                              int K, int rows) {
  unsigned long long ga = (unsigned long long)(size_t)gptr;
  v4u g0;
  g0.x = 1u;                                           // count=1, user mode
  g0.y = lds_addr;                                     // lds_addr (bytes)
  g0.z = (unsigned)ga;                                 // global_addr[31:0]
  g0.w = (unsigned)((ga >> 32) & 0x01FFFFFFu)          // global_addr[56:32]
       | (2u << 30);                                   // type = 2 ("image")
  v8i g1;
  g1[0] = (1 << 16)                                    // data_size = 2 bytes
        | (1 << 20)                                    // pad_enable
        | (4 << 22)                                    // pad_interval: 32 DW
        | (3 << 25);                                   // pad_amount: 4 DW
  g1[1] = (64 & 0xFFFF) << 16;                         // tensor_dim0 = 64
  g1[2] = ((rows & 0xFFFF) << 16);                     // tensor_dim1 = rows
  g1[3] = ((rows >> 16) & 0xFFFF) | (64 << 16);        // tile_dim0 = 64
  g1[4] = rows & 0xFFFF;                               // tile_dim1; dim2=0
  g1[5] = K;                                           // dim0_stride low 32
  g1[6] = 0;
  g1[7] = 0;
  v4i z4 = (v4i)(0);
  v8i z8 = (v8i)(0);
  __builtin_amdgcn_tensor_load_to_lds(g0, g1, z4, z4, z8, 0);
}

// ---------------------------------------------------------------------------
// Embedding: x[b*T+t, :] = embd[tok] + posembd[t]
// ---------------------------------------------------------------------------
__global__ __launch_bounds__(256) void embed_kernel(
    const int* __restrict__ tokens, const float* __restrict__ embd,
    const float* __restrict__ pos, float* __restrict__ x) {
  int row = blockIdx.x;              // 0..NTOK-1
  int t   = row & (NT - 1);
  int tok = tokens[row];
  const float* e = embd + (size_t)tok * DM;
  const float* p = pos  + (size_t)t   * DM;
  float* xr = x + (size_t)row * DM;
  for (int i = threadIdx.x; i < DM; i += 256) xr[i] = e[i] + p[i];
}

// ---------------------------------------------------------------------------
// LayerNorm -> f16 output (one block per token row)
// ---------------------------------------------------------------------------
__global__ __launch_bounds__(256) void ln_kernel(
    const float* __restrict__ x, const float* __restrict__ g,
    const float* __restrict__ bb, _Float16* __restrict__ out) {
  __shared__ float s1[8], s2[8], stats[2];
  int row = blockIdx.x;
  const float* xr = x + (size_t)row * DM;
  float s = 0.f, ss = 0.f;
  for (int i = threadIdx.x; i < DM; i += 256) {
    float v = xr[i]; s += v; ss += v * v;
  }
#pragma unroll
  for (int m = 1; m < 32; m <<= 1) {
    s  += __shfl_xor(s,  m, 32);
    ss += __shfl_xor(ss, m, 32);
  }
  int wv = threadIdx.x >> 5;
  if ((threadIdx.x & 31) == 0) { s1[wv] = s; s2[wv] = ss; }
  __syncthreads();
  if (threadIdx.x == 0) {
    float a = 0.f, b2 = 0.f;
#pragma unroll
    for (int i = 0; i < 8; ++i) { a += s1[i]; b2 += s2[i]; }
    float mean = a * (1.0f / DM);
    float var  = b2 * (1.0f / DM) - mean * mean;
    stats[0] = mean;
    stats[1] = rsqrtf(var + 1e-5f);
  }
  __syncthreads();
  float mean = stats[0], rstd = stats[1];
  _Float16* o = out + (size_t)row * DM;
  for (int i = threadIdx.x; i < DM; i += 256)
    o[i] = (_Float16)((xr[i] - mean) * rstd * g[i] + bb[i]);
}

// ---------------------------------------------------------------------------
// Weight conversion: fp32 [K][N] -> f16 [N][K] (tiled 32x32 transpose).
// ---------------------------------------------------------------------------
__global__ __launch_bounds__(256) void convT(
    const float* __restrict__ in, _Float16* __restrict__ out, int K, int N) {
  __shared__ float tile[32][33];
  int n0 = blockIdx.x * 32;
  int k0 = blockIdx.y * 32;
  int tx = threadIdx.x & 31, ty = threadIdx.x >> 5;   // ty 0..7
#pragma unroll
  for (int j = 0; j < 4; ++j) {
    int k = k0 + ty + j * 8;
    tile[ty + j * 8][tx] = in[(size_t)k * N + n0 + tx];
  }
  __syncthreads();
#pragma unroll
  for (int j = 0; j < 4; ++j) {
    int n = n0 + ty + j * 8;
    out[(size_t)n * K + k0 + tx] = (_Float16)tile[tx][ty + j * 8];
  }
}

// Pack Wq|Wk|Wv ([H,D,HD] each, fp32) into one f16 matrix [3*DM][DM] (n-major,
// K-contiguous rows): out[sel*DM + h*64 + d][c] = Wsel[h][c][d].
__global__ __launch_bounds__(256) void conv_qkvT(
    const float* __restrict__ Wq, const float* __restrict__ Wk,
    const float* __restrict__ Wv, _Float16* __restrict__ out) {
  __shared__ float tile[32][33];
  int bx  = blockIdx.x;
  int dt  = bx & 1;            // d tile (0..1)
  int ct  = (bx >> 1) & 31;    // c tile (0..31)
  int h   = (bx >> 6) & 15;
  int sel = bx >> 10;          // 0..2
  const float* W = (sel == 0) ? Wq : ((sel == 1) ? Wk : Wv);
  int tx = threadIdx.x & 31, ty = threadIdx.x >> 5;
#pragma unroll
  for (int j = 0; j < 4; ++j) {
    int c = ct * 32 + ty + j * 8;
    int d = dt * 32 + tx;
    tile[ty + j * 8][tx] = W[((size_t)(h * DM + c)) * HD + d];
  }
  __syncthreads();
#pragma unroll
  for (int j = 0; j < 4; ++j) {
    int d = dt * 32 + ty + j * 8;
    int n = sel * DM + h * HD + d;
    int c = ct * 32 + tx;
    out[(size_t)n * DM + c] = (_Float16)tile[tx][ty + j * 8];
  }
}

// ---------------------------------------------------------------------------
// WMMA GEMM: C[M,N] = A[M,K] (f16, row-major) * Bt[N][K] (f16, K-major).
// TDM-staged, double-buffered LDS. Per stage: wave0 issues 2 tensor loads for
// the NEXT stage, waits TENSORcnt for the current one, barrier, then 8 waves
// run 8 WMMAs each off b128 LDS fragment reads.
// MODE 1: f16 relu(out + bias)   MODE 2: f32 C += out+bias
// MODE 3: f32 C  = out+bias
// MODE 4: fused-QKV epilogue: cols [0,2*DM) -> f16 C row-major;
//         cols [2*DM,3*DM) (the V block) -> vT[b][h][d][t], one b128 per tile.
// ---------------------------------------------------------------------------
#define LD_ST 72   // padded stride in halves (144 B), produced by TDM padding

template <int MODE>
__global__ __launch_bounds__(256) void gemm_wmma(
    const _Float16* __restrict__ A, const _Float16* __restrict__ Bt,
    void* __restrict__ Cp, const float* __restrict__ bias,
    _Float16* __restrict__ vT, int M, int N, int K) {
  __shared__ _Float16 lA[2][128 * LD_ST];
  __shared__ _Float16 lB[2][64 * LD_ST];

  const int tid  = threadIdx.x;
  const int lane = tid & 31;
  const int wave = tid >> 5;
  const int wm   = wave >> 1;            // 0..3 -> m offset 32*wm
  const int wn   = wave & 1;             // 0..1 -> n offset 32*wn
  const int lr   = lane & 15;
  const int hf   = lane >> 4;
  const int m0g  = blockIdx.y * 128;
  const int n0g  = blockIdx.x * 64;

  v8f acc[2][2];
#pragma unroll
  for (int i = 0; i < 2; ++i)
#pragma unroll
    for (int j = 0; j < 2; ++j) acc[i][j] = (v8f)(0.0f);

  const _Float16* Ab  = A  + (size_t)m0g * K;
  const _Float16* Btb = Bt + (size_t)n0g * K;

  // prologue: TDM loads for stage 0 into buffer 0
  if (wave == 0) {
    tdm_load_tile(Ab,  lds_off(&lA[0][0]), K, 128);
    tdm_load_tile(Btb, lds_off(&lB[0][0]), K, 64);
  }

  int cur = 0;
  for (int k0 = 0; k0 < K; k0 += 64) {
    const bool more = (k0 + 64) < K;
    if (wave == 0) {
      if (more) {   // issue next stage into the other buffer
        tdm_load_tile(Ab  + k0 + 64, lds_off(&lA[cur ^ 1][0]), K, 128);
        tdm_load_tile(Btb + k0 + 64, lds_off(&lB[cur ^ 1][0]), K, 64);
        // TDM completes in order: <=2 outstanding means current stage landed
        __builtin_amdgcn_s_wait_tensorcnt(2);
      } else {
        __builtin_amdgcn_s_wait_tensorcnt(0);
      }
    }
    __syncthreads();   // publish: buf[cur] ready

#pragma unroll
    for (int ks = 0; ks < 64; ks += 32) {
      v16h af[2], bf[2];
      const int kb = hf * 8;
#pragma unroll
      for (int i = 0; i < 2; ++i) {
        int m = wm * 32 + i * 16 + lr;
        af[i] = ld8x2(&lA[cur][m * LD_ST + ks + 2 * kb],
                      &lA[cur][m * LD_ST + ks + 2 * kb + 8]);
      }
#pragma unroll
      for (int j = 0; j < 2; ++j) {
        int n = wn * 32 + j * 16 + lr;
        bf[j] = ld16(&lB[cur][n * LD_ST + ks + hf * 16]);
      }
#pragma unroll
      for (int i = 0; i < 2; ++i)
#pragma unroll
        for (int j = 0; j < 2; ++j)
          acc[i][j] = wmma_f16(af[i], bf[j], acc[i][j]);
    }
    __syncthreads();   // all waves done with buf[cur] before TDM refills it
    cur ^= 1;
  }

  // Epilogue. C layout: lane holds col = lane&15, VGPR r holds row r+8*(lane>>4).
#pragma unroll
  for (int i = 0; i < 2; ++i) {
#pragma unroll
    for (int j = 0; j < 2; ++j) {
      int gmb = m0g + wm * 32 + i * 16 + 8 * hf;   // first of 8 consecutive rows
      int gn  = n0g + wn * 32 + j * 16 + lr;
      if (MODE == 4 && gn >= 2 * DM) {
        // V block: write transposed vT[b][h][d][t]; 8 consecutive t -> 1 b128
        int d  = gn - 2 * DM;
        int h2 = d >> 6, dd = d & 63;
        int bq = gmb >> 10, t = gmb & (NT - 1);
        union { _Float16 h8[8]; uint4 u; } pk;
#pragma unroll
        for (int r = 0; r < 8; ++r) pk.h8[r] = (_Float16)acc[i][j][r];
        *(uint4*)&vT[(((size_t)(bq * NH + h2)) * HD + dd) * NT + t] = pk.u;
      } else {
        float bv = (MODE >= 1 && MODE <= 3) ? bias[gn] : 0.0f;
#pragma unroll
        for (int r = 0; r < 8; ++r) {
          float v = acc[i][j][r];
          size_t o = (size_t)(gmb + r) * N + gn;
          if (MODE == 1) {
            ((_Float16*)Cp)[o] = (_Float16)fmaxf(v + bv, 0.0f);
          } else if (MODE == 2) {
            ((float*)Cp)[o] += v + bv;
          } else if (MODE == 3) {
            ((float*)Cp)[o] = v + bv;
          } else {
            ((_Float16*)Cp)[o] = (_Float16)v;   // MODE 4 q/k block
          }
        }
      }
    }
  }
}

// ---------------------------------------------------------------------------
// Flash attention: one wave handles 16 query rows of one (b,h).
// qkv [NTOK, 3*DM] f16 holds Q (cols h*64) and K (cols DM+h*64);
// V comes pre-transposed: vT[b][h][d][t].
// Per 32-key chunk: 4 WMMA (QK^T) + 4 WMMA (PV) + 1 WMMA (row-sums of P).
// ---------------------------------------------------------------------------
#define LP_ST 40

__global__ __launch_bounds__(32) void attn_kernel(
    const _Float16* __restrict__ qkv, const _Float16* __restrict__ vT,
    _Float16* __restrict__ att) {
  __shared__ _Float16 lP[16 * LP_ST];

  const int lane = threadIdx.x & 31;
  const int lr   = lane & 15;
  const int hf   = lane >> 4;
  const int qb   = blockIdx.x & 63;      // T/16
  const int bh   = blockIdx.x >> 6;
  const int b    = bh >> 4;
  const int h    = bh & 15;
  const int t0   = qb * 16;
  const float scale = 0.125f;            // HD^-0.5

  // Q fragments (16x64 -> two 16x32 A-fragments), direct from global
  const _Float16* qp =
      qkv + ((size_t)(b * NT + t0 + lr)) * (3 * DM) + h * HD;
  const int kb = hf * 16;
  v16h qf[2];
  qf[0] = ld8x2(qp + kb,      qp + 8 + kb);
  qf[1] = ld8x2(qp + 32 + kb, qp + 40 + kb);

  const _Float16* vtb = vT + ((size_t)(b * NH + h)) * HD * NT;

  v16h ones;
#pragma unroll
  for (int e = 0; e < 16; ++e) ones[e] = (_Float16)1.0f;

  v8f oc[4];
#pragma unroll
  for (int s = 0; s < 4; ++s) oc[s] = (v8f)(0.0f);
  v8f rs = (v8f)(0.0f);                  // running row-sums (all cols equal)
  float mrow[8];
#pragma unroll
  for (int r = 0; r < 8; ++r) mrow[r] = -1e30f;

  for (int s0 = 0; s0 < t0 + 16; s0 += 32) {
    // ---- scores: S = Q * K^T (two 16x16 tiles over this 32-key chunk)
    v8f sa[2];
#pragma unroll
    for (int slab = 0; slab < 2; ++slab) {
      sa[slab] = (v8f)(0.0f);
      int sc = s0 + slab * 16 + lr;      // key index for this lane (= B col)
      const _Float16* kp =
          qkv + ((size_t)(b * NT + sc)) * (3 * DM) + DM + h * HD + hf * 16;
      v16h kf0 = ld16(kp);               // inner dims 0..31
      v16h kf1 = ld16(kp + 32);          // inner dims 32..63
      sa[slab] = wmma_f16(qf[0], kf0, sa[slab]);
      sa[slab] = wmma_f16(qf[1], kf1, sa[slab]);
    }

    // ---- online softmax over the 32-key chunk (max via shuffles; sums via WMMA)
#pragma unroll
    for (int r = 0; r < 8; ++r) {
      int trow = t0 + r + 8 * hf;
      float v0 = sa[0][r] * scale;
      float v1 = sa[1][r] * scale;
      if (s0 + lr      > trow) v0 = -1e30f;
      if (s0 + 16 + lr > trow) v1 = -1e30f;
      float m = fmaxf(v0, v1);
#pragma unroll
      for (int msk = 1; msk < 16; msk <<= 1) m = fmaxf(m, __shfl_xor(m, msk, 32));
      float mn = fmaxf(mrow[r], m);
      float fr = __expf(mrow[r] - mn);
      float p0 = __expf(v0 - mn);
      float p1 = __expf(v1 - mn);
      mrow[r] = mn;
      rs[r] *= fr;
#pragma unroll
      for (int s = 0; s < 4; ++s) oc[s][r] *= fr;
      int prow = r + 8 * hf;
      lP[prow * LP_ST + lr]      = (_Float16)p0;
      lP[prow * LP_ST + 16 + lr] = (_Float16)p1;
    }
    __syncthreads();   // single wave: orders LDS writes before fragment reads

    // ---- O += P (16x32) * V (32x64); row-sums += P * ones
    v16h pf = ld8x2(&lP[lr * LP_ST + kb], &lP[lr * LP_ST + 8 + kb]);
    rs = wmma_f16(pf, ones, rs);
#pragma unroll
    for (int s = 0; s < 4; ++s) {
      v16h vf = ld16(&vtb[(size_t)(s * 16 + lr) * NT + s0 + hf * 16]);
      oc[s] = wmma_f16(pf, vf, oc[s]);
    }
    __syncthreads();   // protect lP before next chunk overwrites
  }

  // ---- normalize and write out f16 [NTOK, DM]
#pragma unroll
  for (int s = 0; s < 4; ++s) {
#pragma unroll
    for (int r = 0; r < 8; ++r) {
      int trow = t0 + r + 8 * hf;
      float v = oc[s][r] / rs[r];
      att[((size_t)(b * NT + trow)) * DM + h * HD + s * 16 + lr] = (_Float16)v;
    }
  }
}

// ---------------------------------------------------------------------------
// Host-side orchestration
// ---------------------------------------------------------------------------
extern "C" void kernel_launch(void* const* d_in, const int* in_sizes, int n_in,
                              void* d_out, int out_size, void* d_ws, size_t ws_size,
                              hipStream_t stream) {
  const int*   tokens  = (const int*)  d_in[0];
  const float* embd    = (const float*)d_in[1];
  const float* posembd = (const float*)d_in[2];
  const float* Wq      = (const float*)d_in[3];
  const float* Wk      = (const float*)d_in[4];
  const float* Wv      = (const float*)d_in[5];
  const float* Wproj   = (const float*)d_in[6];
  const float* bproj   = (const float*)d_in[7];
  const float* ln1_g   = (const float*)d_in[8];
  const float* ln1_b   = (const float*)d_in[9];
  const float* ln2_g   = (const float*)d_in[10];
  const float* ln2_b   = (const float*)d_in[11];
  const float* W1      = (const float*)d_in[12];
  const float* b1      = (const float*)d_in[13];
  const float* W2      = (const float*)d_in[14];
  const float* b2      = (const float*)d_in[15];
  const float* lnf_g   = (const float*)d_in[16];
  const float* lnf_b   = (const float*)d_in[17];
  const float* Wlm     = (const float*)d_in[18];
  const float* blm     = (const float*)d_in[19];
  float* logits = (float*)d_out;

  // workspace layout (bytes)
  char* ws = (char*)d_ws;
  float*     x    = (float*)ws;                               // 8 MB
  _Float16*  xn   = (_Float16*)(ws + (size_t)8  * 1024*1024); // 4 MB
  _Float16*  qkv  = (_Float16*)(ws + (size_t)12 * 1024*1024); // 12 MB
  _Float16*  att  = (_Float16*)(ws + (size_t)24 * 1024*1024); // 4 MB
  _Float16*  hid  = (_Float16*)(ws + (size_t)28 * 1024*1024); // 16 MB
  _Float16*  wbuf = (_Float16*)(ws + (size_t)44 * 1024*1024); // 64 MB (Wlm^T)
  _Float16*  vT   = (_Float16*)(ws + (size_t)108 * 1024*1024);// 4 MB

  embed_kernel<<<NTOK, 256, 0, stream>>>(tokens, embd, posembd, x);

  const size_t wqkv_l  = (size_t)NH * DM * HD;   // per-layer Wq/Wk/Wv elems
  const size_t wproj_l = (size_t)DM * DM;
  const size_t w1_l    = (size_t)DM * DFF;
  const size_t w2_l    = (size_t)DFF * DM;

  for (int l = 0; l < NL; ++l) {
    // ln1 -> xn
    ln_kernel<<<NTOK, 256, 0, stream>>>(x, ln1_g + l * DM, ln1_b + l * DM, xn);

    // fused QKV GEMM: [2048,1024] x [1024,3072]; V block lands transposed in vT
    conv_qkvT<<<3 * 16 * 32 * 2, 256, 0, stream>>>(
        Wq + l * wqkv_l, Wk + l * wqkv_l, Wv + l * wqkv_l, wbuf);
    gemm_wmma<4><<<dim3(3 * DM / 64, NTOK / 128), 256, 0, stream>>>(
        xn, wbuf, qkv, nullptr, vT, NTOK, 3 * DM, DM);

    // flash attention
    attn_kernel<<<NB * NH * (NT / 16), 32, 0, stream>>>(qkv, vT, att);

    // x += att @ Wproj + bproj
    convT<<<dim3(DM / 32, DM / 32), 256, 0, stream>>>(
        Wproj + l * wproj_l, wbuf, DM, DM);
    gemm_wmma<2><<<dim3(DM / 64, NTOK / 128), 256, 0, stream>>>(
        att, wbuf, x, bproj + l * DM, nullptr, NTOK, DM, DM);

    // ln2 -> xn
    ln_kernel<<<NTOK, 256, 0, stream>>>(x, ln2_g + l * DM, ln2_b + l * DM, xn);

    // hid = relu(xn @ W1 + b1)
    convT<<<dim3(DFF / 32, DM / 32), 256, 0, stream>>>(
        W1 + l * w1_l, wbuf, DM, DFF);
    gemm_wmma<1><<<dim3(DFF / 64, NTOK / 128), 256, 0, stream>>>(
        xn, wbuf, hid, b1 + l * DFF, nullptr, NTOK, DFF, DM);

    // x += hid @ W2 + b2
    convT<<<dim3(DM / 32, DFF / 32), 256, 0, stream>>>(
        W2 + l * w2_l, wbuf, DFF, DM);
    gemm_wmma<2><<<dim3(DM / 64, NTOK / 128), 256, 0, stream>>>(
        hid, wbuf, x, b2 + l * DM, nullptr, NTOK, DM, DFF);
  }

  // final LN + LM head
  ln_kernel<<<NTOK, 256, 0, stream>>>(x, lnf_g, lnf_b, xn);
  convT<<<dim3(NV / 32, DM / 32), 256, 0, stream>>>(Wlm, wbuf, DM, NV);
  gemm_wmma<3><<<dim3(NV / 64, NTOK / 128), 256, 0, stream>>>(
      xn, wbuf, logits, blm, nullptr, NTOK, NV, DM);
}
